// CostVolume_21114059227573
// MI455X (gfx1250) — matrix operand order repaired and checked by reading, not compiled, not run on hardware
//
#include <hip/hip_runtime.h>

// -------- types --------
typedef __attribute__((ext_vector_type(16))) __bf16 v16bf;
typedef __attribute__((ext_vector_type(8)))  __bf16 v8bf;
typedef __attribute__((ext_vector_type(8)))  float  v8f;
typedef __attribute__((ext_vector_type(4)))  float  v4f;

#define B_DIM 4
#define C_DIM 128
#define H_DIM 128
#define W_DIM 256
#define LVLS  64
#define ROWP  136   // padded [w][c] row stride in bf16 elems (128 + 8): 16B-aligned rows, no bank conflicts
#define IPAD  16    // guard band offset in cost rows
#define CROWS 96    // cost rows cover i in [-16, 80): all tile writes land in-bounds unconditionally

// f32 -> packed 2x bf16 (round-to-nearest-even-ish)
static __device__ __forceinline__ unsigned pack_bf16(float a, float b) {
    unsigned ua = __float_as_uint(a);
    unsigned ub = __float_as_uint(b);
    ua += 0x7fffu + ((ua >> 16) & 1u);
    ub += 0x7fffu + ((ub >> 16) & 1u);
    return (ua >> 16) | (ub & 0xffff0000u);
}

// 16 bf16 as two 16B ds_load_b128; second half at +off elements
static __device__ __forceinline__ v16bf load16(const __bf16* p, int off) {
    union { v16bf v; v8bf h[2]; } u;
    u.h[0] = *(const v8bf*)(p);
    u.h[1] = *(const v8bf*)(p + off);
    return u.v;
}

__global__ __launch_bounds__(256)
void cost_volume_wmma_bf16(const float* __restrict__ f1,
                           const float* __restrict__ f2,
                           float* __restrict__ out)
{
    __shared__ __bf16 f1t[W_DIM * ROWP];     // f1 transposed [w][c], bf16, pre-scaled by 1/128
    __shared__ __bf16 f2t[W_DIM * ROWP];     // f2 transposed [w][c], bf16
    __shared__ float  cost[CROWS * W_DIM];   // guard-banded staging for unconditional scatter

    const int tid  = threadIdx.x;
    const int lane = tid & 31;
    const int wv   = tid >> 5;                 // wave id 0..7

    const int bh = blockIdx.x;                 // 0..511 == (b,h)
    const int b  = bh >> 7;
    const int h  = bh & (H_DIM - 1);

    const float* f1g = f1 + ((size_t)(b * C_DIM) * H_DIM + h) * W_DIM;
    const float* f2g = f2 + ((size_t)(b * C_DIM) * H_DIM + h) * W_DIM;

    // ---- zero only the 64 output rows (w < i region must stay 0) ----
    {
        v4f* cz = (v4f*)(cost + IPAD * W_DIM);
        #pragma unroll
        for (int it = 0; it < 16; ++it) {
            v4f z = {0.f, 0.f, 0.f, 0.f};
            cz[it * 256 + tid] = z;
        }
    }

    // ---- stage f1/f2 rows into LDS, transposed + bf16; fold mean (1/128, exact) into f1 ----
    {
        const int w = tid;                       // thread owns column w
        #pragma unroll 4
        for (int cp = 0; cp < C_DIM / 2; ++cp) {
            const size_t o0 = (size_t)(2 * cp + 0) * (H_DIM * W_DIM) + w;
            const size_t o1 = (size_t)(2 * cp + 1) * (H_DIM * W_DIM) + w;
            unsigned p1 = pack_bf16(f1g[o0] * 0.0078125f, f1g[o1] * 0.0078125f);
            unsigned p2 = pack_bf16(f2g[o0], f2g[o1]);
            *(unsigned*)&f1t[w * ROWP + 2 * cp] = p1;
            *(unsigned*)&f2t[w * ROWP + 2 * cp] = p2;
        }
    }
    __syncthreads();

    // ---- banded GEMM: corr[j, w] = sum_c f2[c,j] * f1[c,w] (A = f2 tile, B = f1 tile, K = c) ----
    const int n   = lane & 15;           // tile column (both matrices)
    const int kA  = (lane >> 4) * 8;     // A K-half select
    const int kB  = (lane >> 4) * 8;     // B K-half select
    const int mhi = (lane >> 4) * 8;     // D row-half select

    #pragma unroll
    for (int half = 0; half < 2; ++half) {
        const int tw = wv + 8 * half;    // w-tile owned by this wave
        const int w0 = tw * 16;
        const int ww = w0 + n;

        // preload B fragments (f1 column tile) for all 4 K-chunks; resident across j-tiles
        v16bf Bfrag[4];
        #pragma unroll
        for (int kc = 0; kc < 4; ++kc)
            Bfrag[kc] = load16(&f1t[ww * ROWP + kc * 32 + kB], 16);

        // fixed 5 j-tiles per w-tile; clamped duplicates rewrite identical values (harmless)
        int j0s[5];
        #pragma unroll
        for (int s = 0; s < 5; ++s) {
            int tj = tw - 4 + s;
            j0s[s] = (tj < 0 ? 0 : tj) * 16;
        }

        // double-buffered A fragments: next tile's 8 ds_load_b128 overlap current 4 WMMAs
        v16bf Abuf0[4], Abuf1[4];
        #pragma unroll
        for (int kc = 0; kc < 4; ++kc)
            Abuf0[kc] = load16(&f2t[(j0s[0] + n) * ROWP + kc * 32 + kA], 16);

        #pragma unroll
        for (int s = 0; s < 5; ++s) {
            v16bf* cur = (s & 1) ? Abuf1 : Abuf0;
            v16bf* nxt = (s & 1) ? Abuf0 : Abuf1;
            if (s < 4) {
                #pragma unroll
                for (int kc = 0; kc < 4; ++kc)
                    nxt[kc] = load16(&f2t[(j0s[s + 1] + n) * ROWP + kc * 32 + kA], 16);
            }

            v8f acc = {0.f, 0.f, 0.f, 0.f, 0.f, 0.f, 0.f, 0.f};
            #pragma unroll
            for (int kc = 0; kc < 4; ++kc)
                acc = __builtin_amdgcn_wmma_f32_16x16x32_bf16(
                        false, cur[kc], false, Bfrag[kc],
                        (short)0, acc, false, false);

            // unconditional scatter into guard-banded cost[i+IPAD][w];
            // per-lane dword addr stride 257 -> conflict-free banks
            const int base = (ww + IPAD - j0s[s] - mhi) * W_DIM + ww;
            #pragma unroll
            for (int v = 0; v < 8; ++v)
                cost[base - v * W_DIM] = acc[v];
        }
    }
    __syncthreads();

    // ---- coalesced b128 stores of the valid [64 x 256] slab (rows IPAD..IPAD+63) ----
    float* og = out + ((size_t)(b * LVLS) * H_DIM + h) * W_DIM;   // base at i=0
    const v4f* cs = (const v4f*)(cost + IPAD * W_DIM);
    #pragma unroll
    for (int it = 0; it < 16; ++it) {
        const int v  = it * 256 + tid;       // vec4 id, 4096 total
        const int i  = v >> 6;               // 64 vec4 per i-row
        const int w4 = (v & 63) * 4;
        v4f val = cs[v];
        *(v4f*)&og[(size_t)i * (H_DIM * W_DIM) + w4] = val;
    }
}

extern "C" void kernel_launch(void* const* d_in, const int* in_sizes, int n_in,
                              void* d_out, int out_size, void* d_ws, size_t ws_size,
                              hipStream_t stream) {
    const float* f1  = (const float*)d_in[0];
    const float* f2  = (const float*)d_in[1];
    float*       out = (float*)d_out;
    (void)in_sizes; (void)n_in; (void)d_ws; (void)ws_size; (void)out_size;

    dim3 grid(B_DIM * H_DIM);   // one workgroup per (b, h)
    dim3 block(256);            // 8 wave32 waves
    cost_volume_wmma_bf16<<<grid, block, 0, stream>>>(f1, f2, out);
}
